// Transformer_85856396247759
// MI455X (gfx1250) — matrix operand order repaired and checked by reading, not compiled, not run on hardware
//
#include <hip/hip_runtime.h>
#include <hip/hip_bf16.h>

// ---------------------------------------------------------------------------
// Transformer forward for MI455X (gfx1250): all GEMMs via v_wmma_f32_16x16x32_f16
// (wave32, 16x16 output tile per wave, fp32 accumulate, f16 convert in regs).
// Round 2: branch-free guards + float4 fast path (previous round's per-element
// guards lowered to divergent exec-mask branches around every scalar load).
// ---------------------------------------------------------------------------

typedef __attribute__((ext_vector_type(16))) _Float16 v16h;
typedef __attribute__((ext_vector_type(8)))  float    v8f;

#define BB   16
#define LL   100
#define DD   512
#define ADIM 64
#define HH   8
#define HIDD 2048
#define VV   32000
#define LPAD 112   // score row pitch (7 tiles of 16)

struct GemmParams {
    const float* A; int lda; long long oA; long long iA;          // A: M x K (row major)
    const float* B; int ldb; int bT;  long long oB; long long iB; // B: K x N (or N x K if bT)
    const float* bias;                                            // [N] or null
    const float* Res;                                             // residual, indexed like C, or null
    float* C; int ldc; int cs; long long oC; long long iC;        // C element (m,n) at m*ldc + n*cs
    int M, N, K;
    float alpha;                                                  // C = alpha*(A@B) + bias + Res
    int relu;
    int Hd;                                                       // inner-batch modulus (heads)
};

// One 16x16 C tile per wave; 8 waves / 256-thread block; grid.y = batch.
__global__ void gemm_wmma_f16(GemmParams p) {
    const int wave = threadIdx.x >> 5;
    const int lane = threadIdx.x & 31;
    const int tilesM = (p.M + 15) >> 4;
    const int tilesN = (p.N + 15) >> 4;
    const int tile = blockIdx.x * (blockDim.x >> 5) + wave;
    if (tile >= tilesM * tilesN) return;          // wave-uniform: EXEC stays all-ones
    const int tm = tile % tilesM;
    const int tn = tile / tilesM;

    const int bz = blockIdx.y;
    const int ob = bz / p.Hd, ib = bz % p.Hd;
    const float* A = p.A + ob * p.oA + ib * p.iA;
    const float* B = p.B + ob * p.oB + ib * p.iB;
    float*       C = p.C + ob * p.oC + ib * p.iC;
    const float* R = p.Res ? (p.Res + ob * p.oC + ib * p.iC) : (const float*)0;

    const int mloc = lane & 15;          // A row within tile / C column within tile
    const int hi   = lane >> 4;          // lane half-group
    const int m = tm * 16 + mloc;        // A load row
    const int n = tn * 16 + mloc;        // B load column

    v8f acc = {};
    // wave-uniform fast/slow decision (EXEC stays all-ones through the WMMAs)
    const bool full = ((tm + 1) * 16 <= p.M) && ((tn + 1) * 16 <= p.N) && ((p.K & 31) == 0);

    if (full) {
        const float* Arow = A + (long long)m * p.lda + hi * 16;
        if (p.bT) {
            const float* Brow = B + (long long)n * p.ldb + hi * 16;
            for (int k0 = 0; k0 < p.K; k0 += 32) {
                v16h a, b;
                float4 f0 = *(const float4*)(Arow + k0);
                float4 f1 = *(const float4*)(Arow + k0 + 4);
                float4 f2 = *(const float4*)(Arow + k0 + 8);
                float4 f3 = *(const float4*)(Arow + k0 + 12);
                a[0]=(_Float16)f0.x; a[1]=(_Float16)f0.y; a[2]=(_Float16)f0.z; a[3]=(_Float16)f0.w;
                a[4]=(_Float16)f1.x; a[5]=(_Float16)f1.y; a[6]=(_Float16)f1.z; a[7]=(_Float16)f1.w;
                a[8]=(_Float16)f2.x; a[9]=(_Float16)f2.y; a[10]=(_Float16)f2.z; a[11]=(_Float16)f2.w;
                a[12]=(_Float16)f3.x; a[13]=(_Float16)f3.y; a[14]=(_Float16)f3.z; a[15]=(_Float16)f3.w;
                float4 g0 = *(const float4*)(Brow + k0);
                float4 g1 = *(const float4*)(Brow + k0 + 4);
                float4 g2 = *(const float4*)(Brow + k0 + 8);
                float4 g3 = *(const float4*)(Brow + k0 + 12);
                b[0]=(_Float16)g0.x; b[1]=(_Float16)g0.y; b[2]=(_Float16)g0.z; b[3]=(_Float16)g0.w;
                b[4]=(_Float16)g1.x; b[5]=(_Float16)g1.y; b[6]=(_Float16)g1.z; b[7]=(_Float16)g1.w;
                b[8]=(_Float16)g2.x; b[9]=(_Float16)g2.y; b[10]=(_Float16)g2.z; b[11]=(_Float16)g2.w;
                b[12]=(_Float16)g3.x; b[13]=(_Float16)g3.y; b[14]=(_Float16)g3.z; b[15]=(_Float16)g3.w;
                acc = __builtin_amdgcn_wmma_f32_16x16x32_f16(false, a, false, b,
                                                             (short)0, acc, false, false);
            }
        } else {
            const float* Bcol = B + n + (long long)hi * 16 * p.ldb;
            for (int k0 = 0; k0 < p.K; k0 += 32) {
                v16h a, b;
                float4 f0 = *(const float4*)(Arow + k0);
                float4 f1 = *(const float4*)(Arow + k0 + 4);
                float4 f2 = *(const float4*)(Arow + k0 + 8);
                float4 f3 = *(const float4*)(Arow + k0 + 12);
                a[0]=(_Float16)f0.x; a[1]=(_Float16)f0.y; a[2]=(_Float16)f0.z; a[3]=(_Float16)f0.w;
                a[4]=(_Float16)f1.x; a[5]=(_Float16)f1.y; a[6]=(_Float16)f1.z; a[7]=(_Float16)f1.w;
                a[8]=(_Float16)f2.x; a[9]=(_Float16)f2.y; a[10]=(_Float16)f2.z; a[11]=(_Float16)f2.w;
                a[12]=(_Float16)f3.x; a[13]=(_Float16)f3.y; a[14]=(_Float16)f3.z; a[15]=(_Float16)f3.w;
                const float* bp = Bcol + (long long)k0 * p.ldb;
                #pragma unroll
                for (int h = 0; h < 16; ++h)
                    b[h] = (_Float16)bp[(long long)h * p.ldb];
                acc = __builtin_amdgcn_wmma_f32_16x16x32_f16(false, a, false, b,
                                                             (short)0, acc, false, false);
            }
        }
    } else {
        // branch-free guarded path: unconditional loads from clamped addresses,
        // then select-to-zero (v_cndmask, no exec-mask branches)
        const bool mok = (m < p.M);
        const bool nok = (n < p.N);
        const int  mc  = mok ? m : (p.M - 1);
        const int  nc  = nok ? n : (p.N - 1);
        const float* Arow = A + (long long)mc * p.lda;
        for (int k0 = 0; k0 < p.K; k0 += 32) {
            v16h a, b;
            #pragma unroll
            for (int v = 0; v < 8; ++v) {
                int k  = k0 + hi * 16 + 2 * v;
                int kc = (k + 1 < p.K) ? k : (p.K - 2);       // K is even in all uses
                float2 f = *(const float2*)(Arow + kc);
                a[2 * v]     = (mok && k     < p.K) ? (_Float16)f.x : (_Float16)0.f;
                a[2 * v + 1] = (mok && k + 1 < p.K) ? (_Float16)f.y : (_Float16)0.f;
            }
            #pragma unroll
            for (int h = 0; h < 16; ++h) {
                int k  = k0 + hi * 16 + h;
                int kc = (k < p.K) ? k : (p.K - 1);
                long long off = p.bT ? ((long long)nc * p.ldb + kc)
                                     : ((long long)kc * p.ldb + nc);
                float x = B[off];
                b[h] = (nok && k < p.K) ? (_Float16)x : (_Float16)0.f;
            }
            acc = __builtin_amdgcn_wmma_f32_16x16x32_f16(false, a, false, b,
                                                         (short)0, acc, false, false);
        }
    }

    // C/D layout: VGPR v holds element (M = v + 8*hi, N = lane&15)
    #pragma unroll
    for (int v = 0; v < 8; ++v) {
        int mm = tm * 16 + v + hi * 8;
        int nn = tn * 16 + mloc;
        if (mm < p.M && nn < p.N) {
            float val = p.alpha * acc[v];
            if (p.bias) val += p.bias[nn];
            long long off = (long long)mm * p.ldc + (long long)nn * p.cs;
            if (R) val += R[off];
            if (p.relu && val < 0.f) val = 0.f;
            C[off] = val;
        }
    }
}

// out[b,l,d] = share_weight[tok[b,l], d] + PE(l, d)
__global__ void embed_pe(const int* __restrict__ tok, const float* __restrict__ sw,
                         float* __restrict__ out, int rows) {
    int idx = blockIdx.x * blockDim.x + threadIdx.x;
    int total = rows * DD;
    if (idx >= total) return;
    int d   = idx & (DD - 1);
    int row = idx >> 9;
    int l   = row % LL;
    int t   = tok[row];
    float e = (t >= 0 && t < VV) ? sw[(long long)t * DD + d] : 0.f;
    float expo = (float)(d >> 1) * (1.0f / (float)DD);
    float ang  = (float)l * __expf(-expo * 9.210340371976184f);   // 10000^-expo
    float pe   = (d & 1) ? __cosf(ang) : __sinf(ang);
    out[idx] = e + pe;
}

// wave-per-row LayerNorm over 512
__global__ void layernorm512(const float* __restrict__ x, const float* __restrict__ g,
                             const float* __restrict__ b, float* __restrict__ y, int rows) {
    int wave = threadIdx.x >> 5, lane = threadIdx.x & 31;
    int row = blockIdx.x * (blockDim.x >> 5) + wave;
    if (row >= rows) return;
    const float* xr = x + (long long)row * DD;
    float v[16];
    float s = 0.f;
    #pragma unroll
    for (int j = 0; j < 16; ++j) { v[j] = xr[lane + 32 * j]; s += v[j]; }
    #pragma unroll
    for (int o = 16; o > 0; o >>= 1) s += __shfl_xor(s, o, 32);
    float mean = s * (1.f / DD);
    float q = 0.f;
    #pragma unroll
    for (int j = 0; j < 16; ++j) { float d = v[j] - mean; q += d * d; }
    #pragma unroll
    for (int o = 16; o > 0; o >>= 1) q += __shfl_xor(q, o, 32);
    float inv = rsqrtf(q * (1.f / DD) + 1e-5f);
    float* yr = y + (long long)row * DD;
    #pragma unroll
    for (int j = 0; j < 16; ++j) {
        int d = lane + 32 * j;
        yr[d] = (v[j] - mean) * inv * g[d] + b[d];
    }
}

// scores layout (B,H,LL,LPAD): set -1e9 where key masked (tok==0, optionally causal)
__global__ void mask_scores(float* __restrict__ S, const int* __restrict__ tok, int causal) {
    int idx = blockIdx.x * blockDim.x + threadIdx.x;
    int total = BB * HH * LL * LL;
    if (idx >= total) return;
    int k  = idx % LL;
    int r  = idx / LL;          // (b*H + h)*LL + l
    int l  = r % LL;
    int b  = r / (LL * HH);
    bool valid = (tok[b * LL + k] != 0);
    if (causal) valid = valid && (k <= l);
    if (!valid) S[(long long)r * LPAD + k] = -1e9f;
}

// wave-per-row softmax over 100 keys (row pitch LPAD)
__global__ void softmax_rows100(float* __restrict__ S, int rows) {
    int wave = threadIdx.x >> 5, lane = threadIdx.x & 31;
    int row = blockIdx.x * (blockDim.x >> 5) + wave;
    if (row >= rows) return;
    float* r = S + (long long)row * LPAD;
    float v[4];
    float mx = -3.4e38f;
    #pragma unroll
    for (int j = 0; j < 4; ++j) {
        int k = lane + 32 * j;
        v[j] = (k < LL) ? r[k] : -3.4e38f;
        mx = fmaxf(mx, v[j]);
    }
    #pragma unroll
    for (int o = 16; o > 0; o >>= 1) mx = fmaxf(mx, __shfl_xor(mx, o, 32));
    float s = 0.f;
    #pragma unroll
    for (int j = 0; j < 4; ++j) {
        int k = lane + 32 * j;
        v[j] = (k < LL) ? __expf(v[j] - mx) : 0.f;
        s += v[j];
    }
    #pragma unroll
    for (int o = 16; o > 0; o >>= 1) s += __shfl_xor(s, o, 32);
    float inv = 1.f / s;
    #pragma unroll
    for (int j = 0; j < 4; ++j) {
        int k = lane + 32 * j;
        if (k < LL) r[k] = v[j] * inv;
    }
}

// block-per-row in-place softmax over 32000 logits
__global__ void softmax_vocab(float* __restrict__ X, int cols) {
    __shared__ float red[256];
    int row = blockIdx.x;
    float* r = X + (long long)row * cols;
    float mx = -3.4e38f;
    for (int c = threadIdx.x; c < cols; c += blockDim.x) mx = fmaxf(mx, r[c]);
    red[threadIdx.x] = mx; __syncthreads();
    for (int o = 128; o > 0; o >>= 1) {
        if ((int)threadIdx.x < o) red[threadIdx.x] = fmaxf(red[threadIdx.x], red[threadIdx.x + o]);
        __syncthreads();
    }
    mx = red[0]; __syncthreads();
    float s = 0.f;
    for (int c = threadIdx.x; c < cols; c += blockDim.x) s += __expf(r[c] - mx);
    red[threadIdx.x] = s; __syncthreads();
    for (int o = 128; o > 0; o >>= 1) {
        if ((int)threadIdx.x < o) red[threadIdx.x] += red[threadIdx.x + o];
        __syncthreads();
    }
    float inv = 1.f / red[0];
    for (int c = threadIdx.x; c < cols; c += blockDim.x) r[c] = __expf(r[c] - mx) * inv;
}

// ---------------------------------------------------------------------------
// host side
// ---------------------------------------------------------------------------

static void run_gemm(hipStream_t s,
                     const float* A, int lda, long long oA, long long iA,
                     const float* B, int ldb, int bT, long long oB, long long iB,
                     const float* bias, const float* res,
                     float* C, int ldc, int cs, long long oC, long long iC,
                     int M, int N, int K, float alpha, int relu, int batch, int Hd) {
    GemmParams p{A, lda, oA, iA, B, ldb, bT, oB, iB, bias, res,
                 C, ldc, cs, oC, iC, M, N, K, alpha, relu, Hd};
    int tiles = ((M + 15) / 16) * ((N + 15) / 16);
    dim3 grid((unsigned)((tiles + 7) / 8), (unsigned)batch, 1);
    gemm_wmma_f16<<<grid, 256, 0, s>>>(p);
}

extern "C" void kernel_launch(void* const* d_in, const int* in_sizes, int n_in,
                              void* d_out, int out_size, void* d_ws, size_t ws_size,
                              hipStream_t stream) {
    (void)in_sizes; (void)n_in; (void)out_size; (void)ws_size;

    const int*   inputs  = (const int*)d_in[0];
    const int*   outputs = (const int*)d_in[1];
    const float* sw      = (const float*)d_in[2];
    const float* fcb     = (const float*)d_in[3];
    const float* W[32];
    for (int i = 0; i < 32; ++i) W[i] = (const float*)d_in[4 + i];
    // per prefix (enc = W[0..15], dec = W[16..31]):
    // 0 wq, 1 bq, 2 wk, 3 bk, 4 wv, 5 bv, 6 wo, 7 bo, 8 g1, 9 b1,
    // 10 f1w, 11 f1b, 12 f2w, 13 f2b, 14 g2, 15 b2

    const int ROWS = BB * LL;                 // 1600
    float* ws = (float*)d_ws;
    float* encA  = ws;               ws += (long long)ROWS * DD;
    float* decA  = ws;               ws += (long long)ROWS * DD;
    float* qb    = ws;               ws += (long long)ROWS * DD;
    float* kb    = ws;               ws += (long long)ROWS * DD;
    float* vb    = ws;               ws += (long long)ROWS * DD;
    float* attn  = ws;               ws += (long long)ROWS * DD;   // head-concat output
    float* preln = ws;               ws += (long long)ROWS * DD;
    float* ffh   = ws;               ws += (long long)ROWS * HIDD;
    float* sc    = ws;               ws += (long long)BB * HH * LL * LPAD;

    const long long BSTR = (long long)LL * DD;        // per-batch activation stride
    const long long SSTR = (long long)HH * LL * LPAD; // per-batch score stride
    const float ISCALE = 0.125f;                      // 1/sqrt(AD)

    // mha: x = LN( attn_out(x->Q, kv->K/V) @ wo + bo + x )
    auto mha = [&](float* x, const float* kv, const float* const* w,
                   const int* mtok, int causal) {
        // Q, K, V projections (1600 x 512 = (1600x512) @ (512x512) + bias)
        run_gemm(stream, x,  DD, 0, 0, w[0], DD, 0, 0, 0, w[1], 0,
                 qb, DD, 1, 0, 0, ROWS, DD, DD, 1.f, 0, 1, 1);
        run_gemm(stream, kv, DD, 0, 0, w[2], DD, 0, 0, 0, w[3], 0,
                 kb, DD, 1, 0, 0, ROWS, DD, DD, 1.f, 0, 1, 1);
        run_gemm(stream, kv, DD, 0, 0, w[4], DD, 0, 0, 0, w[5], 0,
                 vb, DD, 1, 0, 0, ROWS, DD, DD, 1.f, 0, 1, 1);
        // scores = (Q/8) @ K^T, batched over (b,h): 100x100, K=64
        run_gemm(stream, qb, DD, BSTR, ADIM, kb, DD, 1, BSTR, ADIM, 0, 0,
                 sc, LPAD, 1, SSTR, (long long)LL * LPAD,
                 LL, LL, ADIM, ISCALE, 0, BB * HH, HH);
        {
            int total = BB * HH * LL * LL;
            mask_scores<<<(total + 255) / 256, 256, 0, stream>>>(sc, mtok, causal);
            softmax_rows100<<<(BB * HH * LL + 7) / 8, 256, 0, stream>>>(sc, BB * HH * LL);
        }
        // attn @ V, batched; output element (l,d) -> attn[(b*L+l)*512 + d*H + h]
        run_gemm(stream, sc, LPAD, SSTR, (long long)LL * LPAD,
                 vb, DD, 0, BSTR, ADIM, 0, 0,
                 attn, DD, HH, BSTR, 1,
                 LL, ADIM, LL, 1.f, 0, BB * HH, HH);
        // output projection + residual, then LN back into x
        run_gemm(stream, attn, DD, 0, 0, w[6], DD, 0, 0, 0, w[7], x,
                 preln, DD, 1, 0, 0, ROWS, DD, DD, 1.f, 0, 1, 1);
        layernorm512<<<(ROWS + 7) / 8, 256, 0, stream>>>(preln, w[8], w[9], x, ROWS);
    };

    auto ff = [&](float* x, const float* const* w) {
        run_gemm(stream, x, DD, 0, 0, w[10], HIDD, 0, 0, 0, w[11], 0,
                 ffh, HIDD, 1, 0, 0, ROWS, HIDD, DD, 1.f, 1 /*relu*/, 1, 1);
        run_gemm(stream, ffh, HIDD, 0, 0, w[12], DD, 0, 0, 0, w[13], x,
                 preln, DD, 1, 0, 0, ROWS, DD, HIDD, 1.f, 0, 1, 1);
        layernorm512<<<(ROWS + 7) / 8, 256, 0, stream>>>(preln, w[14], w[15], x, ROWS);
    };

    const float* const* encW = &W[0];
    const float* const* decW = &W[16];

    // ------------------------- encoder -------------------------
    embed_pe<<<(ROWS * DD + 255) / 256, 256, 0, stream>>>(inputs, sw, encA, ROWS);
    for (int it = 0; it < 2; ++it) {            // RECYCLE + 1
        mha(encA, encA, encW, inputs, 0);
        ff(encA, encW);
    }

    // ------------------------- decoder -------------------------
    embed_pe<<<(ROWS * DD + 255) / 256, 256, 0, stream>>>(outputs, sw, decA, ROWS);
    for (int it = 0; it < 2; ++it) {
        mha(decA, decA, decW, outputs, 1);      // masked self-attention
        mha(decA, encA, decW, inputs, 0);       // cross-attention
        ff(decA, decW);
    }

    // ----------------- logits (tied weights) + softmax ----------
    // logits = (dec * D^-0.5) @ sw^T + fc_b, written straight into d_out
    float* outp = (float*)d_out;
    run_gemm(stream, decA, DD, 0, 0, sw, DD, 1 /*bT*/, 0, 0, fcb, 0,
             outp, VV, 1, 0, 0, ROWS, VV, DD, 0.044194173824159216f /*512^-0.5*/, 0, 1, 1);
    softmax_vocab<<<ROWS, 256, 0, stream>>>(outp, VV);
}
